// SGC_1005022347292
// MI455X (gfx1250) — matrix operand order repaired and checked by reading, not compiled, not run on hardware
//
#include <hip/hip_runtime.h>

// ---------- types ----------
typedef __attribute__((ext_vector_type(16))) __bf16        v16bf;
typedef __attribute__((ext_vector_type(8)))  float         v8f;
typedef __attribute__((ext_vector_type(4)))  unsigned int  u32x4;
typedef __attribute__((ext_vector_type(4)))  float         f32x4;

union Frag { u32x4 q[2]; v16bf v; };

__device__ inline __bf16 f2bf(float f) {
  unsigned u = __builtin_bit_cast(unsigned, f);
  unsigned r = u + 0x7fffu + ((u >> 16) & 1u);     // round-to-nearest-even
  unsigned short h = (unsigned short)(r >> 16);
  return __builtin_bit_cast(__bf16, h);
}

// ---------- utility kernels ----------
__global__ void zero_k(f32x4* __restrict__ p, long long n4) {
  long long i = (long long)blockIdx.x * blockDim.x + threadIdx.x;
  if (i < n4) p[i] = (f32x4)0.0f;
}

__global__ void deg_k(const int* __restrict__ dst, float* __restrict__ deg, int E) {
  int i = blockIdx.x * blockDim.x + threadIdx.x;
  if (i < E) atomicAdd(&deg[dst[i]], 1.0f);
}

__global__ void norm_k(float* __restrict__ deg, int N) {
  int i = blockIdx.x * blockDim.x + threadIdx.x;
  if (i < N) deg[i] = rsqrtf(fmaxf(deg[i], 1.0f));   // in-place deg -> norm
}

// out[dst] += xin[src] * norm[src]^pw  ;  one float4 chunk per thread (64 chunks/row)
__global__ void scatter_k(const float* __restrict__ xin, const int* __restrict__ src,
                          const int* __restrict__ dst, const float* __restrict__ nrm,
                          float* __restrict__ xout, long long total, int pw) {
  long long idx = (long long)blockIdx.x * blockDim.x + threadIdx.x;
  if (idx >= total) return;
  int e = (int)(idx >> 6);
  int j = (int)(idx & 63);
  int s = src[e];
  int d = dst[e];
  float sc = nrm[s];
  if (pw == 2) sc *= sc;
  f32x4 v = ((const f32x4*)(xin + (size_t)s * 256))[j];
  float* o = xout + (size_t)d * 256 + (size_t)j * 4;
  atomicAdd(o + 0, v.x * sc);
  atomicAdd(o + 1, v.y * sc);
  atomicAdd(o + 2, v.z * sc);
  atomicAdd(o + 3, v.w * sc);
}

// bf16 staging for GEMM input, fused with post-hop norm scaling: out = bf16(xin * norm[row])
__global__ void conv_k(const float* __restrict__ xin, const float* __restrict__ nrm,
                       __bf16* __restrict__ out, long long total) {
  long long i = (long long)blockIdx.x * blockDim.x + threadIdx.x;
  if (i < total) out[i] = f2bf(xin[i] * nrm[i >> 8]);
}

// Pack W (K=256 rows x ncols) into WMMA B-fragment order:
// frag(tile t, kstep s): lane holds col = t*16 + lane%16, elements i -> K = s*32 + 16*(lane/16) + i
__global__ void pack_w(const float* __restrict__ W, int ncols, int tiles, __bf16* __restrict__ Wp) {
  int idx = blockIdx.x * blockDim.x + threadIdx.x;
  int total = tiles * 8 * 32 * 16;
  if (idx >= total) return;
  int i    = idx & 15;
  int lane = (idx >> 4) & 31;
  int s    = (idx >> 9) & 7;
  int t    = idx >> 12;
  int k = s * 32 + 16 * (lane >> 4) + i;
  int c = t * 16 + (lane & 15);
  float v = (c < ncols) ? W[(size_t)k * ncols + c] : 0.0f;
  Wp[idx] = f2bf(v);
}

// ---------- WMMA GEMM: Y[N x ldY] = relu?(Xb[N x 256] * W + bias) ----------
// block = 256 threads = 8 waves; wave w -> rows [by*128 + w*16, +16), cols [bx*16, +16)
__global__ void gemm_bf16(const __bf16* __restrict__ Xb, const __bf16* __restrict__ Wp,
                          const float* __restrict__ bias, float* __restrict__ Y,
                          int N, int ncolsValid, int ldY, int doRelu) {
  int lane = threadIdx.x & 31;
  int wave = threadIdx.x >> 5;
  int rowbase = blockIdx.y * 128 + wave * 16;
  if (rowbase >= N) return;                      // N % 16 == 0: whole-tile guard
  int half = lane >> 4;
  int l15  = lane & 15;
  int col  = blockIdx.x * 16 + l15;

  float bv = (col < ncolsValid) ? bias[col] : 0.0f;
  v8f acc;
#pragma unroll
  for (int i = 0; i < 8; ++i) acc[i] = bv;

  const __bf16* arow = Xb + (size_t)(rowbase + l15) * 256;
  const __bf16* bbase = Wp + ((size_t)blockIdx.x * 8 * 32) * 16;

#pragma unroll
  for (int s = 0; s < 8; ++s) {
    int kbase = s * 32;
    Frag a, b;
    a.q[0] = *(const u32x4*)(arow + kbase + 16 * half);       // K = 16h .. 16h+7
    a.q[1] = *(const u32x4*)(arow + kbase + 16 * half + 8);   // K = 16h+8 .. 16h+15
    const __bf16* bp = bbase + ((size_t)s * 32 + lane) * 16;  // 32B contiguous per lane
    b.q[0] = ((const u32x4*)bp)[0];
    b.q[1] = ((const u32x4*)bp)[1];
    acc = __builtin_amdgcn_wmma_f32_16x16x32_bf16(
        /*neg_a=*/false, a.v, /*neg_b=*/false, b.v,
        /*c_mod=*/(short)0, acc, /*reuse_a=*/false, /*reuse_b=*/false);
  }

  if (doRelu) {
#pragma unroll
    for (int i = 0; i < 8; ++i) acc[i] = fmaxf(acc[i], 0.0f);
  }
  if (col < ncolsValid) {
#pragma unroll
    for (int i = 0; i < 8; ++i) {
      int r = rowbase + half * 8 + i;                         // C/D layout: M = i + 8*half
      Y[(size_t)r * ldY + col] = acc[i];
    }
  }
}

// ---------- host ----------
static inline unsigned cdiv(long long a, long long b) { return (unsigned)((a + b - 1) / b); }

extern "C" void kernel_launch(void* const* d_in, const int* in_sizes, int n_in,
                              void* d_out, int out_size, void* d_ws, size_t ws_size,
                              hipStream_t stream) {
  const float* feat = (const float*)d_in[0];
  const int*   src  = (const int*)d_in[1];
  const int*   dst  = (const int*)d_in[2];
  const float* W1   = (const float*)d_in[3];
  const float* b1   = (const float*)d_in[4];
  const float* W2   = (const float*)d_in[5];
  const float* b2   = (const float*)d_in[6];
  const float* W3   = (const float*)d_in[7];
  const float* b3   = (const float*)d_in[8];
  float* out = (float*)d_out;

  const int N = in_sizes[0] / 256;
  const int E = in_sizes[1];
  const long long NF = (long long)N * 256;
  const long long ET = (long long)E * 64;   // (edge, float4-chunk) pairs

  char* w = (char*)d_ws;
  auto carve = [&](size_t bytes) { char* p = w; w += (bytes + 255) & ~(size_t)255; return p; };
  float*  A   = (float*)carve((size_t)NF * 4);
  float*  B   = (float*)carve((size_t)NF * 4);
  __bf16* Xb  = (__bf16*)carve((size_t)NF * 2);
  float*  nrm = (float*)carve((size_t)N * 4);
  __bf16* Wp1 = (__bf16*)carve((size_t)16 * 8 * 32 * 16 * 2);
  __bf16* Wp2 = (__bf16*)carve((size_t)16 * 8 * 32 * 16 * 2);
  __bf16* Wp3 = (__bf16*)carve((size_t)3 * 8 * 32 * 16 * 2);

  const dim3 blk(256);
  const unsigned gN4  = cdiv(N, 4 * 256);        // zero norm (N/4 float4s)
  const unsigned gNF4 = cdiv(NF / 4, 256);       // zero N x 256
  const unsigned gE   = cdiv(E, 256);
  const unsigned gN   = cdiv(N, 256);
  const unsigned gET  = cdiv(ET, 256);
  const unsigned gNF  = cdiv(NF, 256);
  const dim3 gemmGrid256(16, cdiv(N, 128));
  const dim3 gemmGrid40(3, cdiv(N, 128));

  // degree -> norm
  zero_k<<<gN4, blk, 0, stream>>>((f32x4*)nrm, N / 4);
  deg_k<<<gE, blk, 0, stream>>>(dst, nrm, E);
  norm_k<<<gN, blk, 0, stream>>>(nrm, N);

  // weight packing (tiny, once per call)
  pack_w<<<cdiv(16 * 4096, 256), blk, 0, stream>>>(W1, 256, 16, Wp1);
  pack_w<<<cdiv(16 * 4096, 256), blk, 0, stream>>>(W2, 256, 16, Wp2);
  pack_w<<<cdiv(3 * 4096, 256), blk, 0, stream>>>(W3, 40, 3, Wp3);

  // layer 1: h1 = relu(prop(x,1) @ W1 + b1)
  zero_k<<<gNF4, blk, 0, stream>>>((f32x4*)A, NF / 4);
  scatter_k<<<gET, blk, 0, stream>>>(feat, src, dst, nrm, A, ET, 1);
  conv_k<<<gNF, blk, 0, stream>>>(A, nrm, Xb, NF);
  gemm_bf16<<<gemmGrid256, blk, 0, stream>>>(Xb, Wp1, b1, B, N, 256, 256, 1);

  // layer 2: h2 = relu(prop(h1,2) @ W2 + b2)
  zero_k<<<gNF4, blk, 0, stream>>>((f32x4*)A, NF / 4);
  scatter_k<<<gET, blk, 0, stream>>>(B, src, dst, nrm, A, ET, 1);
  zero_k<<<gNF4, blk, 0, stream>>>((f32x4*)B, NF / 4);
  scatter_k<<<gET, blk, 0, stream>>>(A, src, dst, nrm, B, ET, 2);  // norm^2: post-hop1 * pre-hop2
  conv_k<<<gNF, blk, 0, stream>>>(B, nrm, Xb, NF);
  gemm_bf16<<<gemmGrid256, blk, 0, stream>>>(Xb, Wp2, b2, A, N, 256, 256, 1);

  // layer 3: out = prop(h2,1) @ W3 + b3
  zero_k<<<gNF4, blk, 0, stream>>>((f32x4*)B, NF / 4);
  scatter_k<<<gET, blk, 0, stream>>>(A, src, dst, nrm, B, ET, 1);
  conv_k<<<gNF, blk, 0, stream>>>(B, nrm, Xb, NF);
  gemm_bf16<<<gemmGrid40, blk, 0, stream>>>(Xb, Wp3, b3, out, N, 40, 40, 0);
}
